// ElmanLayer_58145267254151
// MI455X (gfx1250) — matrix operand order repaired and checked by reading, not compiled, not run on hardware
//
#include <hip/hip_runtime.h>
#include <hip/hip_bf16.h>
#include <math.h>

// Problem dims (fixed by the reference)
#define BB 64
#define TT 512
#define II 512
#define HH 1024

typedef __attribute__((ext_vector_type(16))) __bf16 v16bf;
typedef __attribute__((ext_vector_type(8)))  float  v8f;

// ---------------------------------------------------------------------------
// Pack W (row-major [N][K], f32) into WMMA B-fragment order (bf16):
//   tiles are (nt, kt) with kt-major inner; within a 32x16 tile each lane
//   owns two 8-wide K runs of one column:
//     b[i] = W[n0 + lane%16][k0 + (i>=8?16:0) + (lane>=16?8:0) + i%8]
//   stored contiguously: tile*1024B + lane*32B + i*2B.
// ---------------------------------------------------------------------------
__global__ __launch_bounds__(256) void pack_w_kernel(
    const float* __restrict__ W, __bf16* __restrict__ out, int N, int K) {
  int idx = blockIdx.x * blockDim.x + threadIdx.x;   // one bf16 element each
  if (idx >= N * K) return;
  int ntk     = K >> 5;                 // tiles along K
  int tile    = idx >> 9;               // 512 elements per tile
  int within  = idx & 511;
  int lane    = within >> 4;
  int i       = within & 15;
  int nt      = tile / ntk;
  int kt      = tile - nt * ntk;
  int n       = (nt << 4) + (lane & 15);
  int k       = (kt << 5) + ((i >> 3) << 4) + ((lane >> 4) << 3) + (i & 7);
  out[idx] = (__bf16)W[(size_t)n * K + k];
}

// Convert one 16x32 f32 A-fragment chunk (two contiguous 8-float runs).
__device__ __forceinline__ v16bf make_a_frag(const float* p) {
  v8f a0 = *(const v8f*)(p);        // K = klo .. klo+7
  v8f a1 = *(const v8f*)(p + 16);   // K = klo+16 .. klo+23
  v16bf a;
#pragma unroll
  for (int i = 0; i < 8; ++i) {
    a[i]     = (__bf16)a0[i];
    a[i + 8] = (__bf16)a1[i];
  }
  return a;
}

// ---------------------------------------------------------------------------
// Phase 1: xW[m,n] = input[m,:] . W_ih[n,:] + b_ih[n]   (M = B*T rows)
// Written straight into d_out (flattened [B,T,H] == [M,H]); the scan kernel
// later overwrites each slice in place with tanh(xW + h@Whh^T).
// One 16x16 tile per wave; 8 waves/block share the same A (m) tile.
// Two independent accumulators break the WMMA RAW chain.
// ---------------------------------------------------------------------------
__global__ __launch_bounds__(256) void gemm_xw_kernel(
    const float* __restrict__ in, const __bf16* __restrict__ Bp,
    const float* __restrict__ bias, float* __restrict__ out) {
  const int lane  = threadIdx.x & 31;
  const int wave  = threadIdx.x >> 5;
  const int mt    = blockIdx.x >> 3;                    // 0..2047
  const int nt    = ((blockIdx.x & 7) << 3) + wave;     // 0..63
  const int l16   = lane & 15;
  const int lhalf = lane >> 4;
  const int klo   = lhalf << 3;                         // 0 or 8
  const int m     = (mt << 4) + l16;                    // A row for this lane

  const float*  arow  = in + (size_t)m * II + klo;
  const __bf16* bbase = Bp + (size_t)nt * (II / 32) * 512 + lane * 16;

  v8f acc0 = {}, acc1 = {};
#pragma unroll 2
  for (int kk = 0; kk < II / 32; kk += 2) {
    v16bf a0 = make_a_frag(arow + kk * 32);
    v16bf b0 = *(const v16bf*)(bbase + kk * 512);
    acc0 = __builtin_amdgcn_wmma_f32_16x16x32_bf16(
        false, a0, false, b0, (short)0, acc0, false, false);
    v16bf a1 = make_a_frag(arow + (kk + 1) * 32);
    v16bf b1 = *(const v16bf*)(bbase + (kk + 1) * 512);
    acc1 = __builtin_amdgcn_wmma_f32_16x16x32_bf16(
        false, a1, false, b1, (short)0, acc1, false, false);
  }

  const int n  = (nt << 4) + l16;
  const float bi = bias[n];
  float* obase = out + (size_t)((mt << 4) + (lhalf << 3)) * HH + n;
#pragma unroll
  for (int r = 0; r < 8; ++r)
    obase[(size_t)r * HH] = acc0[r] + acc1[r] + bi;
}

// ---------------------------------------------------------------------------
// Phase 2, one timestep: for all (b,n),
//   d_out[b,t,n] = tanh( d_out[b,t,n] + sum_k d_out[b,t-1,k] * Whh[n,k] )
// 256 tiles (4 m-tiles x 64 n-tiles), 8 waves/block, 32 blocks.
// t==0: h_prev = 0, skip the GEMM.
// ---------------------------------------------------------------------------
__global__ __launch_bounds__(256) void rnn_step_kernel(
    const __bf16* __restrict__ Whhp, float* out, int t) {
  const int lane  = threadIdx.x & 31;
  const int wave  = threadIdx.x >> 5;
  const int tile  = (blockIdx.x << 3) + wave;   // 0..255
  const int nt    = tile >> 2;                  // 0..63
  const int mt    = tile & 3;                   // 0..3
  const int l16   = lane & 15;
  const int lhalf = lane >> 4;
  const int klo   = lhalf << 3;
  const int m     = (mt << 4) + l16;

  v8f acc0 = {}, acc1 = {};
  if (t > 0) {
    const float*  hrow  = out + (size_t)m * (TT * HH) + (size_t)(t - 1) * HH + klo;
    const __bf16* bbase = Whhp + (size_t)nt * (HH / 32) * 512 + lane * 16;
#pragma unroll 2
    for (int kk = 0; kk < HH / 32; kk += 2) {
      v16bf a0 = make_a_frag(hrow + kk * 32);
      v16bf b0 = *(const v16bf*)(bbase + kk * 512);
      acc0 = __builtin_amdgcn_wmma_f32_16x16x32_bf16(
          false, a0, false, b0, (short)0, acc0, false, false);
      v16bf a1 = make_a_frag(hrow + (kk + 1) * 32);
      v16bf b1 = *(const v16bf*)(bbase + (kk + 1) * 512);
      acc1 = __builtin_amdgcn_wmma_f32_16x16x32_bf16(
          false, a1, false, b1, (short)0, acc1, false, false);
    }
  }

  const int n = (nt << 4) + l16;
#pragma unroll
  for (int r = 0; r < 8; ++r) {
    size_t oidx = (size_t)((mt << 4) + (lhalf << 3) + r) * (TT * HH)
                + (size_t)t * HH + n;
    out[oidx] = tanhf(acc0[r] + acc1[r] + out[oidx]);
  }
}

// ---------------------------------------------------------------------------
extern "C" void kernel_launch(void* const* d_in, const int* in_sizes, int n_in,
                              void* d_out, int out_size, void* d_ws, size_t ws_size,
                              hipStream_t stream) {
  (void)in_sizes; (void)n_in; (void)out_size; (void)ws_size;
  const float* input = (const float*)d_in[0];  // [B,T,I]
  const float* W_ih  = (const float*)d_in[1];  // [H,I]
  const float* b_ih  = (const float*)d_in[2];  // [H]
  const float* W_hh  = (const float*)d_in[3];  // [H,H]
  float* out = (float*)d_out;                  // [B,T,H]

  __bf16* wih_p = (__bf16*)d_ws;               // H*I  bf16 = 1 MB
  __bf16* whh_p = wih_p + (size_t)HH * II;     // H*H  bf16 = 2 MB

  // Pack weights into WMMA B-fragment order (cheap one-shot kernels).
  pack_w_kernel<<<(HH * II) / 256, 256, 0, stream>>>(W_ih, wih_p, HH, II);
  pack_w_kernel<<<(HH * HH) / 256, 256, 0, stream>>>(W_hh, whh_p, HH, HH);

  // Phase 1: xW + bias into d_out.  M tiles = 2048, N tile-groups = 8.
  gemm_xw_kernel<<<2048 * 8, 256, 0, stream>>>(input, wih_p, b_ih, out);

  // Phase 2: sequential scan; stream order provides the dependency.
  for (int t = 0; t < TT; ++t)
    rnn_step_kernel<<<32, 256, 0, stream>>>(whh_p, out, t);
}
